// RL_AP_GCN_29824252903502
// MI455X (gfx1250) — matrix operand order repaired and checked by reading, not compiled, not run on hardware
//
#include <hip/hip_runtime.h>
#include <hip/hip_bf16.h>
#include <math.h>

typedef __attribute__((ext_vector_type(16))) _Float16 v16h;
typedef __attribute__((ext_vector_type(8)))  float    v8f;
typedef __attribute__((ext_vector_type(2)))  __fp16   h2v;   // matches cvt_pkrtz return type

#define D_IN 256
#define HID  64
#define CC   64
#define NITER 10
#define EPSV 1e-10f
#define EXPL 0.1f

// ---------------------------------------------------------------- utilities
__device__ __forceinline__ unsigned hashu(unsigned a, unsigned b) {
    unsigned h = a * 0x9E3779B9u ^ (b * 0x85EBCA6Bu) ^ 1234u; // RNG_SALT
    h ^= h >> 16; h *= 0x7FEB352Du;
    h ^= h >> 15; h *= 0x846CA68Bu;
    h ^= h >> 16;
    return h;
}
__device__ __forceinline__ float u01(unsigned h) {
    return (float)(h >> 8) * (1.0f / 16777216.0f) + 1e-7f;
}

// ------------------------------------------------- weight f32 -> f16 convert
__global__ void cvt_weights(const float* __restrict__ W0, const float* __restrict__ W1,
                            _Float16* __restrict__ W0h, _Float16* __restrict__ W1h) {
    int i = blockIdx.x * blockDim.x + threadIdx.x;
    if (i < D_IN * HID) W0h[i] = (_Float16)W0[i];
    else if (i < D_IN * HID + HID * CC) {
        int j = i - D_IN * HID;
        W1h[j] = (_Float16)W1[j];
    }
}

// ------------------------------------------------------------- state init
__global__ void init_state(int* __restrict__ deg, int* __restrict__ act,
                           float* __restrict__ steps, float* __restrict__ hlp,
                           float* __restrict__ hv, float* __restrict__ he, int n) {
    int i = blockIdx.x * blockDim.x + threadIdx.x;
    if (i < n) {
        deg[i] = 0; act[i] = 1;
        steps[i] = 1.0f; hlp[i] = 0.0f; hv[i] = 0.0f; he[i] = 0.0f;
    }
}

__global__ void deg_accum(const int* __restrict__ dst, int* __restrict__ deg, int e) {
    int i = blockIdx.x * blockDim.x + threadIdx.x;
    if (i < e) atomicAdd(&deg[dst[i]], 1);
}

__global__ void dinv_k(const int* __restrict__ deg, float* __restrict__ dinv, int n) {
    int i = blockIdx.x * blockDim.x + threadIdx.x;
    if (i < n) dinv[i] = rsqrtf((float)deg[i] + 1.0f);   // +1: self loop
}

__global__ void norm_k(const int* __restrict__ src, const int* __restrict__ dst,
                       const float* __restrict__ dinv, float* __restrict__ nrm, int e) {
    int i = blockIdx.x * blockDim.x + threadIdx.x;
    if (i < e) nrm[i] = dinv[src[i]] * dinv[dst[i]];
}

// ------------------------------------------- feature MLP with WMMA (f16 x f16 -> f32)
// block = 128 threads (4 waves). Each block handles 16-row tiles (grid-stride).
// Wave w owns output columns [16w, 16w+16).
__global__ __launch_bounds__(128) void feat_mlp(
    const float* __restrict__ x, const _Float16* __restrict__ W0h,
    const float* __restrict__ b0, const _Float16* __restrict__ W1h,
    const float* __restrict__ b1, float* __restrict__ prop, int n, int nTiles)
{
    __shared__ _Float16 xt[16 * D_IN];   // 8 KB
    __shared__ _Float16 h1t[16 * HID];   // 2 KB

    const int tid   = threadIdx.x;
    const int lane  = tid & 31;
    const int wid   = tid >> 5;
    const int n0    = wid * 16;          // column base for this wave
    const int mHalf = lane >> 4;         // K sub-block selector
    const int nCol  = lane & 15;

    // --- hoist B fragments (weights) out of the tile loop -------------------
    v16h b0f[8];
#pragma unroll
    for (int kc = 0; kc < 8; ++kc) {
        const int kb = kc * 32;
#pragma unroll
        for (int e = 0; e < 16; ++e)
            b0f[kc][e] = W0h[(kb + ((e >> 3) << 4) + (mHalf << 3) + (e & 7)) * HID + n0 + nCol];
    }
    v16h b1f[2];
#pragma unroll
    for (int kc = 0; kc < 2; ++kc) {
        const int kb = kc * 32;
#pragma unroll
        for (int e = 0; e < 16; ++e)
            b1f[kc][e] = W1h[(kb + ((e >> 3) << 4) + (mHalf << 3) + (e & 7)) * CC + n0 + nCol];
    }
    const float bias0 = b0[n0 + nCol];
    const float bias1 = b1[n0 + nCol];

    for (int tile = blockIdx.x; tile < nTiles; tile += gridDim.x) {
        const long long row0 = (long long)tile * 16;
        const bool full = (row0 + 16 <= (long long)n);

        // prefetch next tile of x (speculative global_prefetch_b8)
        if (tile + gridDim.x < nTiles)
            __builtin_prefetch((const void*)(x + (row0 + (long long)gridDim.x * 16) * D_IN + tid), 0, 3);

        // stage x tile to LDS as f16
        if (full) {
            // vectorized: 1024 float4 loads / block -> cvt_pk_rtz -> 8B LDS stores
            const float4* xs = (const float4*)(x + row0 * D_IN);
#pragma unroll
            for (int it = 0; it < 8; ++it) {
                const int i4 = it * 128 + tid;        // float4 index 0..1023
                float4 v = xs[i4];
                h2v lo = __builtin_amdgcn_cvt_pkrtz(v.x, v.y);
                h2v hi = __builtin_amdgcn_cvt_pkrtz(v.z, v.w);
                h2v* dp = (h2v*)(xt + (size_t)i4 * 4);
                dp[0] = lo; dp[1] = hi;
            }
        } else {
            for (int i = tid; i < 16 * D_IN; i += 128) {
                int r = i / D_IN;
                float vv = ((row0 + r) < n) ? x[row0 * D_IN + i] : 0.0f;
                xt[i] = (_Float16)vv;
            }
        }
        __syncthreads();

        // GEMM1: [16,256] x [256,64] -> wave's [16,16] slice, K in 8 chunks of 32
        v8f acc = {};
#pragma unroll
        for (int kc = 0; kc < 8; ++kc) {
            v16h a;
            const int kb = kc * 32;
#pragma unroll
            for (int e = 0; e < 16; ++e) {
                const int K = kb + ((e >> 3) << 4) + (mHalf << 3) + (e & 7);
                a[e] = xt[(lane & 15) * D_IN + K];
            }
            acc = __builtin_amdgcn_wmma_f32_16x16x32_f16(
                false, a, false, b0f[kc], (short)0, acc, false, false);
        }
        // bias + relu -> LDS (f16) for second GEMM
#pragma unroll
        for (int r = 0; r < 8; ++r) {
            const int M = mHalf * 8 + r;
            h1t[M * HID + n0 + nCol] = (_Float16)fmaxf(acc[r] + bias0, 0.0f);
        }
        __syncthreads();

        // GEMM2: [16,64] x [64,64], K in 2 chunks of 32
        v8f acc2 = {};
#pragma unroll
        for (int kc = 0; kc < 2; ++kc) {
            v16h a;
            const int kb = kc * 32;
#pragma unroll
            for (int e = 0; e < 16; ++e) {
                const int K = kb + ((e >> 3) << 4) + (mHalf << 3) + (e & 7);
                a[e] = h1t[(lane & 15) * HID + K];
            }
            acc2 = __builtin_amdgcn_wmma_f32_16x16x32_f16(
                false, a, false, b1f[kc], (short)0, acc2, false, false);
        }
        if (full) {
#pragma unroll
            for (int r = 0; r < 8; ++r) {
                const int M = mHalf * 8 + r;
                prop[(row0 + M) * CC + n0 + nCol] = acc2[r] + bias1;
            }
        } else {
#pragma unroll
            for (int r = 0; r < 8; ++r) {
                const int M = mHalf * 8 + r;
                if (row0 + M < n)
                    prop[(row0 + M) * CC + n0 + nCol] = acc2[r] + bias1;
            }
        }
        __syncthreads();
    }
}

// ------------------------------------- propagate: self-loop init + edge scatter
__global__ void self_init(const float* __restrict__ dinv, const float* __restrict__ prop,
                          float* __restrict__ newp, int n) {
    int i = blockIdx.x * blockDim.x + threadIdx.x;
    if (i < n * CC) {
        int node = i >> 6;
        float di = dinv[node];
        newp[i] = di * di * prop[i];
    }
}

// 16 lanes per edge, float4 gather + 4 atomic f32 adds (L2 RMW; working set is L2-resident)
__global__ void edge_prop(const int* __restrict__ src, const int* __restrict__ dst,
                          const float* __restrict__ nrm, const float* __restrict__ prop,
                          float* __restrict__ newp, int e) {
    unsigned tid = blockIdx.x * blockDim.x + threadIdx.x;
    int ei = tid >> 4;
    int part = tid & 15;
    if (ei >= e) return;
    int s = src[ei], d = dst[ei];
    float w = nrm[ei];
    const float4* ps = (const float4*)(prop + (long long)s * CC);
    float4 v = ps[part];
    float* pd = newp + (long long)d * CC + part * 4;
    atomicAdd(pd + 0, w * v.x);
    atomicAdd(pd + 1, w * v.y);
    atomicAdd(pd + 2, w * v.z);
    atomicAdd(pd + 3, w * v.w);
}

// ---------------------------------- per-node ACT step: prop update + policy/value MLPs
// block = 256 (8 waves, one node per wave)
__global__ __launch_bounds__(256) void step_k(
    float* __restrict__ prop, const float* __restrict__ newp, int* __restrict__ act,
    float* __restrict__ steps, float* __restrict__ hlp, float* __restrict__ hv,
    float* __restrict__ he,
    const float* __restrict__ pw1, const float* __restrict__ pb1,
    const float* __restrict__ pw2, const float* __restrict__ pb2,
    const float* __restrict__ pw3, const float* __restrict__ pb3,
    const float* __restrict__ vw1, const float* __restrict__ vb1,
    const float* __restrict__ vw2, const float* __restrict__ vb2,
    const float* __restrict__ vw3, const float* __restrict__ vb3,
    int t, int lastFlag, int n)
{
    __shared__ float sPw1[CC * 32], sVw1[CC * 32];      // 2048 each
    __shared__ float sPw2[32 * 16], sVw2[32 * 16];      // 512 each
    __shared__ float sPw3[16], sVw3[16];
    __shared__ float sPb1[32], sVb1[32], sPb2[16], sVb2[16];
    __shared__ float rowb[8 * CC];
    __shared__ float h1p[8 * 32], h1v[8 * 32];
    __shared__ float h2p[8 * 16], h2v_[8 * 16];

    const int tid  = threadIdx.x;
    const int lane = tid & 31;
    const int wid  = tid >> 5;

    for (int i = tid; i < CC * 32; i += 256) { sPw1[i] = pw1[i]; sVw1[i] = vw1[i]; }
    for (int i = tid; i < 32 * 16; i += 256) { sPw2[i] = pw2[i]; sVw2[i] = vw2[i]; }
    if (tid < 16) { sPw3[tid] = pw3[tid]; sVw3[tid] = vw3[tid];
                    sPb2[tid] = pb2[tid]; sVb2[tid] = vb2[tid]; }
    if (tid < 32) { sPb1[tid] = pb1[tid]; sVb1[tid] = vb1[tid]; }

    int node = blockIdx.x * 8 + wid;
    const bool valid = node < n;
    if (!valid) node = n - 1;
    const long long base = (long long)node * CC;
    const bool a = act[node] != 0;

    // prop = active ? new_prop : prop ; stash row in LDS
    float v0, v1;
    if (a) {
        v0 = newp[base + lane * 2];
        v1 = newp[base + lane * 2 + 1];
        if (valid) {
            prop[base + lane * 2]     = v0;
            prop[base + lane * 2 + 1] = v1;
        }
    } else {
        v0 = prop[base + lane * 2];
        v1 = prop[base + lane * 2 + 1];
    }
    rowb[wid * CC + lane * 2]     = v0;
    rowb[wid * CC + lane * 2 + 1] = v1;
    __syncthreads();

    // layer 1 (64 -> 32), lane j computes output j for both heads
    float ap = sPb1[lane], av = sVb1[lane];
#pragma unroll 8
    for (int k = 0; k < CC; ++k) {
        float r = rowb[wid * CC + k];
        ap = fmaf(r, sPw1[k * 32 + lane], ap);
        av = fmaf(r, sVw1[k * 32 + lane], av);
    }
    h1p[wid * 32 + lane] = fmaxf(ap, 0.0f);
    h1v[wid * 32 + lane] = fmaxf(av, 0.0f);
    __syncthreads();

    // layer 2 (32 -> 16)
    {
        const int j = lane & 15;
        float bp = sPb2[j], bv = sVb2[j];
#pragma unroll 8
        for (int k = 0; k < 32; ++k) {
            bp = fmaf(h1p[wid * 32 + k], sPw2[k * 16 + j], bp);
            bv = fmaf(h1v[wid * 32 + k], sVw2[k * 16 + j], bv);
        }
        if (lane < 16) {
            h2p[wid * 16 + lane]  = fmaxf(bp, 0.0f);
            h2v_[wid * 16 + lane] = fmaxf(bv, 0.0f);
        }
    }
    __syncthreads();

    // head (16 -> 1), halting/value bookkeeping on lane 0
    if (lane == 0 && valid) {
        float dp = pb3[0], dv = vb3[0];
#pragma unroll
        for (int k = 0; k < 16; ++k) {
            dp = fmaf(h2p[wid * 16 + k], sPw3[k], dp);
            dv = fmaf(h2v_[wid * 16 + k], sVw3[k], dv);
        }
        float p   = 1.0f / (1.0f + expf(-dp));
        float ent = -(p * logf(p + EPSV) + (1.0f - p) * logf(1.0f - p + EPSV));

        // deterministic per-(node, t) RNG: normal via Box-Muller + uniform
        float uA = u01(hashu((unsigned)node, (unsigned)(t * 3 + 0)));
        float uB = u01(hashu((unsigned)node, (unsigned)(t * 3 + 1)));
        float nrm01 = sqrtf(-2.0f * logf(uA)) * cosf(6.28318530718f * uB);
        float np_ = fminf(fmaxf(p + EXPL * nrm01, 0.01f), 0.99f);
        float u   = u01(hashu((unsigned)node, (unsigned)(t * 3 + 2)));

        if (a) {
            bool halt = u < np_;
            if (halt) {
                hlp[node] = logf(np_ + EPSV);
                hv[node]  = dv;
                he[node]  = ent;
                act[node] = 0;
            } else {
                steps[node] = (float)(t + 2);
                if (lastFlag) {          // never halted: record last-step outputs
                    hlp[node]   = logf(np_ + EPSV);
                    hv[node]    = dv;
                    he[node]    = ent;
                    steps[node] = (float)NITER;
                }
            }
        }
    }
}

// -------------------------------------------------- log_softmax over 64 classes
__global__ __launch_bounds__(256) void lsm_k(const float* __restrict__ prop,
                                             float* __restrict__ out, int n) {
    const int lane = threadIdx.x & 31;
    const int wid  = threadIdx.x >> 5;
    int node = blockIdx.x * 8 + wid;
    if (node >= n) return;
    const long long base = (long long)node * CC;
    float a0 = prop[base + lane * 2];
    float a1 = prop[base + lane * 2 + 1];
    float m = fmaxf(a0, a1);
#pragma unroll
    for (int off = 16; off > 0; off >>= 1) m = fmaxf(m, __shfl_xor(m, off, 32));
    float s = expf(a0 - m) + expf(a1 - m);
#pragma unroll
    for (int off = 16; off > 0; off >>= 1) s += __shfl_xor(s, off, 32);
    float lse = m + logf(s);
    out[base + lane * 2]     = a0 - lse;
    out[base + lane * 2 + 1] = a1 - lse;
}

// ---------------------------------------------------------------------- launch
extern "C" void kernel_launch(void* const* d_in, const int* in_sizes, int n_in,
                              void* d_out, int out_size, void* d_ws, size_t ws_size,
                              hipStream_t stream) {
    const float* x   = (const float*)d_in[0];
    const int*   ei  = (const int*)  d_in[1];
    const float* W0  = (const float*)d_in[2];
    const float* b0  = (const float*)d_in[3];
    const float* W1  = (const float*)d_in[4];
    const float* b1  = (const float*)d_in[5];
    const float* pw1 = (const float*)d_in[6];
    const float* pb1 = (const float*)d_in[7];
    const float* pw2 = (const float*)d_in[8];
    const float* pb2 = (const float*)d_in[9];
    const float* pw3 = (const float*)d_in[10];
    const float* pb3 = (const float*)d_in[11];
    const float* vw1 = (const float*)d_in[12];
    const float* vb1 = (const float*)d_in[13];
    const float* vw2 = (const float*)d_in[14];
    const float* vb2 = (const float*)d_in[15];
    const float* vw3 = (const float*)d_in[16];
    const float* vb3 = (const float*)d_in[17];

    const int N = in_sizes[0] / D_IN;
    const int E = in_sizes[1] / 2;
    const int* src = ei;
    const int* dst = ei + E;

    // workspace carve-out (deterministic; 256B aligned)
    char* w = (char*)d_ws;
    size_t off = 0;
    auto carve = [&](size_t bytes) { char* p = w + off; off = (off + bytes + 255) & ~(size_t)255; return p; };
    _Float16* W0h  = (_Float16*)carve((size_t)D_IN * HID * 2);
    _Float16* W1h  = (_Float16*)carve((size_t)HID * CC * 2);
    float*    prop = (float*)carve((size_t)N * CC * 4);
    float*    newp = (float*)carve((size_t)N * CC * 4);
    int*      deg  = (int*)  carve((size_t)N * 4);
    float*    dinv = (float*)carve((size_t)N * 4);
    float*    nrm  = (float*)carve((size_t)E * 4);
    int*      act  = (int*)  carve((size_t)N * 4);
    (void)ws_size; (void)n_in; (void)out_size;

    float* out   = (float*)d_out;
    float* steps = out + (size_t)N * CC;
    float* hlp   = steps + N;
    float* hv    = hlp + N;
    float* he    = hv + N;

    const int T = 256;
    const int nTiles = (N + 15) / 16;

    cvt_weights<<<(D_IN * HID + HID * CC + T - 1) / T, T, 0, stream>>>(W0, W1, W0h, W1h);
    init_state<<<(N + T - 1) / T, T, 0, stream>>>(deg, act, steps, hlp, hv, he, N);
    deg_accum<<<(E + T - 1) / T, T, 0, stream>>>(dst, deg, E);
    dinv_k<<<(N + T - 1) / T, T, 0, stream>>>(deg, dinv, N);
    norm_k<<<(E + T - 1) / T, T, 0, stream>>>(src, dst, dinv, nrm, E);

    feat_mlp<<<640, 128, 0, stream>>>(x, W0h, b0, W1h, b1, prop, N, nTiles);

    for (int t = 0; t < NITER; ++t) {
        self_init<<<((size_t)N * CC + T - 1) / T, T, 0, stream>>>(dinv, prop, newp, N);
        edge_prop<<<((size_t)E * 16 + T - 1) / T, T, 0, stream>>>(src, dst, nrm, prop, newp, E);
        step_k<<<(N + 7) / 8, 256, 0, stream>>>(prop, newp, act, steps, hlp, hv, he,
                                                pw1, pb1, pw2, pb2, pw3, pb3,
                                                vw1, vb1, vw2, vb2, vw3, vb3,
                                                t, (t == NITER - 1) ? 1 : 0, N);
    }
    lsm_k<<<(N + 7) / 8, 256, 0, stream>>>(prop, out, N);
}
